// TinyLSTMModel_69715909149001
// MI455X (gfx1250) — matrix operand order repaired and checked by reading, not compiled, not run on hardware
//
#include <hip/hip_runtime.h>

typedef __attribute__((ext_vector_type(16))) _Float16 v16h;
typedef __attribute__((ext_vector_type(8)))  float    v8f;
typedef __attribute__((ext_vector_type(4), aligned(4))) float f4a;

namespace {
constexpr int kT = 2048;
constexpr int kB = 4096;
constexpr int kH = 5;
constexpr int kI = 5;
constexpr int kG = 20;        // 4*H gates
constexpr int kK = kI + kH;   // 10 = concat [x | h]

__device__ __forceinline__ float swap_half(float v) {
  // ds_swizzle group-of-32: xor=0x10, or=0, and=0x1f  (SWAPX16)
  return __int_as_float(__builtin_amdgcn_ds_swizzle(__float_as_int(v), 0x401F));
}

__device__ __forceinline__ float sigmoid_f(float x) {
  return __builtin_amdgcn_rcpf(1.0f + __expf(-x));
}

__device__ __forceinline__ float tanh_f(float x) {
  x = fminf(fmaxf(x, -15.0f), 15.0f);
  float e = __expf(2.0f * x);
  return (e - 1.0f) * __builtin_amdgcn_rcpf(e + 1.0f);
}
} // namespace

__global__ __launch_bounds__(256) void lstm_wmma_kernel(
    const float* __restrict__ x, const float* __restrict__ Wih,
    const float* __restrict__ Whh, const float* __restrict__ bih,
    const float* __restrict__ bhh, float* __restrict__ out) {
  const int lane    = threadIdx.x & 31;
  const int wave    = (blockIdx.x * blockDim.x + threadIdx.x) >> 5;
  const int bbase   = wave * 16;            // 16 batch rows per wave
  const int n       = lane & 15;            // batch row within tile (= N col)
  const bool act    = lane < 16;            // lanes 0-15 own the results
  const int half_hi = (lane >> 4) & 1;

  // ---- A fragments: Wcat[g][k] (g = gate row = M, k = K), constant over t.
  // f16 A 16x32 layout (contiguous K, matching B): lanes 0-15 halfs 0..15 -> K=0..15;
  // lanes 16-31: K=16..31. Our K only reaches 9, rest padded with zeros.
  v16h a0, a1;
  #pragma unroll
  for (int p = 0; p < 16; ++p) {
    int K = p + (half_hi ? 16 : 0);
    float w0 = 0.f, w1 = 0.f;
    int g0i = lane & 15;        // tile0: gate rows 0..15
    int g1i = 16 + (lane & 15); // tile1: gate rows 16..19 (rest zero)
    if (K < kK) {
      w0 = (K < kI) ? Wih[g0i * kI + K] : Whh[g0i * kH + (K - kI)];
      if (g1i < kG)
        w1 = (K < kI) ? Wih[g1i * kI + K] : Whh[g1i * kH + (K - kI)];
    }
    a0[p] = (_Float16)w0;
    a1[p] = (_Float16)w1;
  }

  // ---- combined bias rides in the WMMA C operand (free accumulate)
  v8f cb0, cb1;
  #pragma unroll
  for (int r = 0; r < 8; ++r) {
    int gA = r + half_hi * 8;                    // rows 0..15
    cb0[r] = bih[gA] + bhh[gA];
    int gB = 16 + r + half_hi * 8;               // rows 16..31 (valid < 20)
    cb1[r] = (gB < kG) ? (bih[gB] + bhh[gB]) : 0.f;
  }

  // ---- recurrent state: B fragment holds [x|h] column for this lane's batch
  // f16 B 32x16, contiguous K as in A: lanes 0-15 -> N=lane, K = half index 0..15; lanes 16-31
  // carry K=16..31 which is all padding -> keep zero.
  v16h b;
  #pragma unroll
  for (int p = 0; p < 16; ++p) b[p] = (_Float16)0.f;
  float c[kH];
  #pragma unroll
  for (int j = 0; j < kH; ++j) c[j] = 0.f;

  const size_t bg = (size_t)bbase + n;

  for (int t = 0; t < kT; ++t) {
    const float* xt = x + ((size_t)t * kB + bg) * kI;
    f4a  xv = *(const f4a*)xt;   // lanes 16-31 duplicate lane-16's load (masked below)
    float x4 = xt[4];

    // prefetch next timestep's slice of x (global_prefetch_b8)
    const int tn = (t + 1 < kT) ? (t + 1) : t;
    __builtin_prefetch(x + ((size_t)tn * kB + bg) * kI, 0, 1);

    #pragma unroll
    for (int p = 0; p < 4; ++p) b[p] = (_Float16)(act ? xv[p] : 0.f);
    b[4] = (_Float16)(act ? x4 : 0.f);

    // gates^T = Wcat @ [x|h]^T + bias   (two M tiles: rows 0-15 and 16-19)
    v8f g0 = __builtin_amdgcn_wmma_f32_16x16x32_f16(false, a0, false, b,
                                                    (short)0, cb0, false, false);
    v8f g1 = __builtin_amdgcn_wmma_f32_16x16x32_f16(false, a1, false, b,
                                                    (short)0, cb1, false, false);

    // rows M=8..15 live in the opposite 16-lane half; swap them over
    float sw[8];
    #pragma unroll
    for (int r = 0; r < 8; ++r) sw[r] = swap_half(g0[r]);

    float hv[kH];
    #pragma unroll
    for (int j = 0; j < kH; ++j) {
      float pi = g0[j];                              // row j      : i_j
      float pf = (j < 3) ? g0[5 + j] : sw[j - 3];    // row 5 + j  : f_j
      float pg = sw[2 + j];                          // row 10 + j : g_j
      float po = (j == 0) ? sw[7] : g1[j - 1];       // row 15 + j : o_j
      float fi = sigmoid_f(pi);
      float ff = sigmoid_f(pf);
      float gg = tanh_f(pg);
      float oo = sigmoid_f(po);
      c[j]  = ff * c[j] + fi * gg;
      hv[j] = oo * tanh_f(c[j]);
      // recycle h straight into the B fragment for the next timestep
      b[5 + j] = (_Float16)(act ? hv[j] : 0.f);
    }

    if (act) {
      float* op = out + ((size_t)t * kB + bg) * kH;
      f4a hv4 = {hv[0], hv[1], hv[2], hv[3]};
      *(f4a*)op = hv4;
      op[4] = hv[4];
    }
  }
}

extern "C" void kernel_launch(void* const* d_in, const int* in_sizes, int n_in,
                              void* d_out, int out_size, void* d_ws, size_t ws_size,
                              hipStream_t stream) {
  (void)in_sizes; (void)n_in; (void)d_ws; (void)ws_size; (void)out_size;
  const float* x   = (const float*)d_in[0];
  const float* Wih = (const float*)d_in[1];
  const float* Whh = (const float*)d_in[2];
  const float* bih = (const float*)d_in[3];
  const float* bhh = (const float*)d_in[4];
  float* out = (float*)d_out;

  const int waves   = kB / 16;                 // 256 independent waves
  const int threads = 256;                     // 8 waves per block
  const int blocks  = (waves * 32) / threads;  // 32 blocks
  hipLaunchKernelGGL(lstm_wmma_kernel, dim3(blocks), dim3(threads), 0, stream,
                     x, Wih, Whh, bih, bhh, out);
}
